// LinearAttention_31559419691444
// MI455X (gfx1250) — matrix operand order repaired and checked by reading, not compiled, not run on hardware
//
#include <hip/hip_runtime.h>
#include <hip/hip_bf16.h>

// LinearAttention pipeline for gfx1250 (MI455X), all GEMMs via v_wmma_f32_16x16x32_bf16.
// b=8, C=256, L=16384, heads=8, dim_head=32.

#define Bx   8
#define Cc   256
#define Ll   16384
#define O3   768
#define HEADS 8
#define DHEAD 32
#define SCALE 0.17677669529663689f   // 32^-0.5
#define GN_EPS 1e-5f

typedef __attribute__((ext_vector_type(16))) __bf16 v16bf;
typedef __attribute__((ext_vector_type(8)))  __bf16 v8bf;
typedef __attribute__((ext_vector_type(8)))  float  v8f;

// Cheap round-to-nearest (ties away) f32->bf16: one v_add + d16_hi store.
static __device__ __forceinline__ __bf16 f32_to_bf16(float f) {
  union { float f; unsigned u; } v; v.f = f;
  union { unsigned short s; __bf16 b; } o;
  o.s = (unsigned short)((v.u + 0x8000u) >> 16);
  return o.b;
}

// Reuse hints as template bools so the builtin sees ICE arguments.
template <bool RA, bool RB>
static __device__ __forceinline__ v8f wmma_bf16(v16bf a, v16bf b, v8f c) {
  return __builtin_amdgcn_wmma_f32_16x16x32_bf16(
      /*neg_a=*/false, a, /*neg_b=*/false, b,
      /*c_mod=*/(short)0, c, /*reuse_a=*/RA, /*reuse_b=*/RB);
}

static __device__ __forceinline__ v16bf pack16(v8bf lo, v8bf hi) {
  v16bf r;
#pragma unroll
  for (int i = 0; i < 8; ++i) { r[i] = lo[i]; r[i + 8] = hi[i]; }
  return r;
}

// ---------------------------------------------------------------------------
// K1: cast weights to bf16 (kept L2-resident; A-fragments read directly).
// ---------------------------------------------------------------------------
__global__ __launch_bounds__(256) void cast_weights(
    const float* __restrict__ w_qkv, const float* __restrict__ w_out,
    __bf16* __restrict__ wq_bf, __bf16* __restrict__ wo_bf) {
  int i = blockIdx.x * 256 + threadIdx.x;
  if (i < O3 * Cc)  wq_bf[i] = f32_to_bf16(w_qkv[i]);
  if (i < Cc * Cc)  wo_bf[i] = f32_to_bf16(w_out[i]);
}

// ---------------------------------------------------------------------------
// K2: QKV projection GEMM: qkv[b,o,l] = sum_c w[o,c] * x[b,c,l], bf16 out.
// 64(O) x 256(L) tile, 8 waves, each wave an 8-deep WMMA chain (A reused),
// K-loop 256/32 with double-buffered LDS staging, b128 loads, and coalesced
// b128 stores via an LDS output swap.
// ---------------------------------------------------------------------------
__global__ __launch_bounds__(256) void qkv_gemm(
    const float* __restrict__ x, const __bf16* __restrict__ wq,
    __bf16* __restrict__ qkv) {
  const int l0 = blockIdx.x * 256;
  const int o0 = blockIdx.y * 64;
  const int b  = blockIdx.z;
  __shared__ union __align__(16) {
    __bf16 xT[2][256][40];   // [l][c] transposed, 40-half stride (16B-aligned b128)
    __bf16 osm[64][256];     // output swap tile
  } sm;
  const int tid  = threadIdx.x;
  const int wave = tid >> 5, lane = tid & 31;
  const int lm = lane & 15, hi = lane >> 4;
  const int mrow = wave & 3;              // 16-row O strip
  const int ncol = (wave >> 2) * 128;     // 128-wide L slab

  // Staging: 8 float4 loads per thread per K-step (32c x 256l tile).
  const float* xp[8]; int sc[8], sl[8];
#pragma unroll
  for (int i = 0; i < 8; ++i) {
    int u = tid + i * 256;                // 0..2047 float4 units
    sc[i] = u >> 6;                       // c 0..31
    sl[i] = (u & 63) * 4;                 // l 0,4,..,252
    xp[i] = x + (size_t)(b * Cc + sc[i]) * Ll + l0 + sl[i];
  }
  const size_t kstride = (size_t)32 * Ll;

  auto stage = [&](int buf) {
#pragma unroll
    for (int i = 0; i < 8; ++i) {
      float4 v = *(const float4*)xp[i];
      sm.xT[buf][sl[i] + 0][sc[i]] = f32_to_bf16(v.x);
      sm.xT[buf][sl[i] + 1][sc[i]] = f32_to_bf16(v.y);
      sm.xT[buf][sl[i] + 2][sc[i]] = f32_to_bf16(v.z);
      sm.xT[buf][sl[i] + 3][sc[i]] = f32_to_bf16(v.w);
      xp[i] += kstride;
    }
  };

  v8f acc[8] = {};
  const __bf16* wrow = wq + (size_t)(o0 + mrow * 16 + lm) * Cc + hi * 8;

  stage(0);
  __syncthreads();
  for (int k0 = 0; k0 < Cc; k0 += 32) {
    const int cur = (k0 >> 5) & 1;
    if (k0 + 32 < Cc) stage(cur ^ 1);     // prefetch next tile while computing
    v16bf afrag = pack16(*(const v8bf*)(wrow + k0), *(const v8bf*)(wrow + k0 + 16));
#pragma unroll
    for (int nt = 0; nt < 8; ++nt) {
      const __bf16* bp = &sm.xT[cur][ncol + nt * 16 + lm][hi * 8];
      v16bf bfrag = pack16(*(const v8bf*)bp, *(const v8bf*)(bp + 16));
      if (nt == 0) acc[nt] = wmma_bf16<false, false>(afrag, bfrag, acc[nt]);
      else         acc[nt] = wmma_bf16<true,  false>(afrag, bfrag, acc[nt]);
    }
    __syncthreads();
  }

  // Output swap through LDS -> coalesced global_store_b128.
#pragma unroll
  for (int nt = 0; nt < 8; ++nt)
#pragma unroll
    for (int r = 0; r < 8; ++r)
      sm.osm[mrow * 16 + r + hi * 8][ncol + nt * 16 + lm] = f32_to_bf16(acc[nt][r]);
  __syncthreads();
  __bf16* qbase = qkv + (size_t)(b * O3 + o0) * Ll + l0;
#pragma unroll
  for (int i = 0; i < 8; ++i) {
    int u = tid + i * 256;                // 0..2047 v8bf units (64 x 32)
    int o = u >> 5, l8 = (u & 31) * 8;
    *(v8bf*)(qbase + (size_t)o * Ll + l8) = *(const v8bf*)&sm.osm[o][l8];
  }
}

// ---------------------------------------------------------------------------
// K3: softmax over per-head channel dim (d=32) for q, fused * SCALE. In place.
// ---------------------------------------------------------------------------
__global__ __launch_bounds__(256) void q_softmax(__bf16* __restrict__ qkv) {
  size_t idx = (size_t)blockIdx.x * 256 + threadIdx.x;   // (b,h,l)
  int l  = (int)(idx % Ll);
  int bh = (int)(idx / Ll);
  int h = bh & 7, b = bh >> 3;
  __bf16* base = qkv + (size_t)(b * O3 + h * DHEAD) * Ll + l;
  float v[DHEAD], mx = -1e30f;
#pragma unroll 4
  for (int d = 0; d < DHEAD; ++d) { float t = (float)base[(size_t)d * Ll]; v[d] = t; mx = fmaxf(mx, t); }
  float s = 0.f;
#pragma unroll 4
  for (int d = 0; d < DHEAD; ++d) { float e = __expf(v[d] - mx); v[d] = e; s += e; }
  float inv = SCALE / s;
#pragma unroll 4
  for (int d = 0; d < DHEAD; ++d) base[(size_t)d * Ll] = f32_to_bf16(v[d] * inv);
}

// ---------------------------------------------------------------------------
// K4: softmax over L=16384 for k. One WG per row; row staged once into LDS
// (32 KB of the 320 KB WGP pool), all three phases run from ds reads.
// ---------------------------------------------------------------------------
__global__ __launch_bounds__(256) void k_softmax(__bf16* __restrict__ qkv) {
  int b = blockIdx.x >> 8;
  int o = Cc + (blockIdx.x & 255);                 // k region: o in [256,512)
  __bf16* row = qkv + (size_t)(b * O3 + o) * Ll;
  __shared__ __align__(16) __bf16 rowL[Ll];
  __shared__ float red[256];
  int t = threadIdx.x;
#pragma unroll
  for (int i = 0; i < 8; ++i) {                    // b128 copy global -> LDS
    int idx = (t + i * 256) * 8;
    *(v8bf*)&rowL[idx] = *(const v8bf*)(row + idx);
  }
  __syncthreads();
  float mx = -1e30f;
#pragma unroll
  for (int i = 0; i < 8; ++i) {
    v8bf v = *(const v8bf*)&rowL[(t + i * 256) * 8];
#pragma unroll
    for (int j = 0; j < 8; ++j) mx = fmaxf(mx, (float)v[j]);
  }
  red[t] = mx; __syncthreads();
  for (int s = 128; s; s >>= 1) { if (t < s) red[t] = fmaxf(red[t], red[t + s]); __syncthreads(); }
  mx = red[0]; __syncthreads();
  float sum = 0.f;
#pragma unroll
  for (int i = 0; i < 8; ++i) {
    v8bf v = *(const v8bf*)&rowL[(t + i * 256) * 8];
#pragma unroll
    for (int j = 0; j < 8; ++j) sum += __expf((float)v[j] - mx);
  }
  red[t] = sum; __syncthreads();
  for (int s = 128; s; s >>= 1) { if (t < s) red[t] += red[t + s]; __syncthreads(); }
  float inv = 1.0f / red[0];
#pragma unroll
  for (int i = 0; i < 8; ++i) {
    int idx = (t + i * 256) * 8;
    v8bf v = *(const v8bf*)&rowL[idx], w;
#pragma unroll
    for (int j = 0; j < 8; ++j) w[j] = f32_to_bf16(__expf((float)v[j] - mx) * inv);
    *(v8bf*)(row + idx) = w;                       // b128 store back
  }
}

// ---------------------------------------------------------------------------
// K5: context[b,h,d,e] = sum_l k_sm[d,l] * v[e,l].  32x32 GEMM with K=16384.
// One WG per (b,h); 8 waves split K, reduce via LDS ds_add_f32 atomics.
// A/B fragments load directly from global (both contiguous along l).
// ---------------------------------------------------------------------------
__global__ __launch_bounds__(256) void context_gemm(
    const __bf16* __restrict__ qkv, float* __restrict__ ctx) {
  int b = blockIdx.x >> 3, h = blockIdx.x & 7;
  const __bf16* krow = qkv + (size_t)(b * O3 + Cc     + h * DHEAD) * Ll;
  const __bf16* vrow = qkv + (size_t)(b * O3 + 2 * Cc + h * DHEAD) * Ll;
  __shared__ float cred[DHEAD * DHEAD];
  int tid = threadIdx.x, wave = tid >> 5, lane = tid & 31;
  int lm = lane & 15, hi = lane >> 4;
  for (int i = tid; i < DHEAD * DHEAD; i += 256) cred[i] = 0.f;
  __syncthreads();

  const int lbase = wave * (Ll / 8);
  const __bf16* ap0 = krow + (size_t)( 0 + lm) * Ll + lbase + hi * 8;
  const __bf16* ap1 = krow + (size_t)(16 + lm) * Ll + lbase + hi * 8;
  const __bf16* bp0 = vrow + (size_t)( 0 + lm) * Ll + lbase + hi * 8;
  const __bf16* bp1 = vrow + (size_t)(16 + lm) * Ll + lbase + hi * 8;

  v8f acc[2][2] = {};
  for (int k = 0; k < Ll / 8; k += 32) {
    v16bf a0 = pack16(*(const v8bf*)(ap0 + k), *(const v8bf*)(ap0 + k + 16));
    v16bf a1 = pack16(*(const v8bf*)(ap1 + k), *(const v8bf*)(ap1 + k + 16));
    v16bf b0 = pack16(*(const v8bf*)(bp0 + k), *(const v8bf*)(bp0 + k + 16));
    v16bf b1 = pack16(*(const v8bf*)(bp1 + k), *(const v8bf*)(bp1 + k + 16));
    acc[0][0] = wmma_bf16<false, false>(a0, b0, acc[0][0]);
    acc[0][1] = wmma_bf16<true,  false>(a0, b1, acc[0][1]);
    acc[1][0] = wmma_bf16<false, false>(a1, b0, acc[1][0]);
    acc[1][1] = wmma_bf16<true,  false>(a1, b1, acc[1][1]);
  }
#pragma unroll
  for (int mt = 0; mt < 2; ++mt)
#pragma unroll
    for (int nt = 0; nt < 2; ++nt)
#pragma unroll
      for (int r = 0; r < 8; ++r) {
        int d = mt * 16 + r + hi * 8;
        int e = nt * 16 + lm;
        atomicAdd(&cred[d * DHEAD + e], acc[mt][nt][r]);
      }
  __syncthreads();
  float* cg = ctx + (size_t)(b * HEADS + h) * (DHEAD * DHEAD);
  for (int i = tid; i < DHEAD * DHEAD; i += 256) cg[i] = cred[i];
}

// ---------------------------------------------------------------------------
// K6: attn[b,h*32+e,l] = sum_d ctx[d,e] * q_sm[b,h*32+d,l].  K=32 single WMMA.
// ctx^T staged as LDS A-tile; q tile LDS-transposed for B; output swapped
// through LDS for coalesced b128 stores.
// ---------------------------------------------------------------------------
__global__ __launch_bounds__(256) void attn_apply(
    const __bf16* __restrict__ qkv, const float* __restrict__ ctx,
    __bf16* __restrict__ attn) {
  int l0 = blockIdx.x * 256;
  int b = blockIdx.y >> 3, h = blockIdx.y & 7;
  __shared__ union __align__(16) {
    __bf16 qT[256][40];                    // [l][d]
    __bf16 osm[DHEAD][256];                // output swap tile [e][l]
  } sm;
  __shared__ __align__(16) __bf16 aT[DHEAD * DHEAD];  // [e][d]  (A = ctx^T)
  const float*  cg   = ctx + (size_t)(b * HEADS + h) * (DHEAD * DHEAD);
  const __bf16* qrow = qkv + (size_t)(b * O3 + h * DHEAD) * Ll;
  int tid = threadIdx.x;
  for (int i = tid; i < DHEAD * DHEAD; i += 256) {
    int d = i >> 5, e = i & 31;
    aT[e * DHEAD + d] = f32_to_bf16(cg[d * DHEAD + e]);
  }
#pragma unroll
  for (int i = 0; i < 4; ++i) {            // 1024 v8bf units: 32 d x 256 l
    int u = tid + i * 256;
    int d = u >> 5, l8 = (u & 31) * 8;
    v8bf v = *(const v8bf*)(qrow + (size_t)d * Ll + l0 + l8);
#pragma unroll
    for (int j = 0; j < 8; ++j) sm.qT[l8 + j][d] = v[j];
  }
  __syncthreads();
  int wave = tid >> 5, lane = tid & 31, lm = lane & 15, hi = lane >> 4;
  int nbase = wave * 32;
  v16bf afr[2], bfr[2];
#pragma unroll
  for (int mt = 0; mt < 2; ++mt) {
    const __bf16* ap = &aT[(mt * 16 + lm) * DHEAD + hi * 8];
    afr[mt] = pack16(*(const v8bf*)ap, *(const v8bf*)(ap + 16));
  }
#pragma unroll
  for (int nt = 0; nt < 2; ++nt) {
    const __bf16* bp = &sm.qT[nbase + nt * 16 + lm][hi * 8];
    bfr[nt] = pack16(*(const v8bf*)bp, *(const v8bf*)(bp + 16));
  }
  v8f c00 = {}, c01 = {}, c10 = {}, c11 = {};
  c00 = wmma_bf16<false, false>(afr[0], bfr[0], c00);
  c01 = wmma_bf16<true,  false>(afr[0], bfr[1], c01);
  c10 = wmma_bf16<false, false>(afr[1], bfr[0], c10);
  c11 = wmma_bf16<true,  false>(afr[1], bfr[1], c11);
  __syncthreads();                         // all qT reads done; reuse as osm
  v8f* cs[4] = { &c00, &c01, &c10, &c11 };
#pragma unroll
  for (int mt = 0; mt < 2; ++mt)
#pragma unroll
    for (int nt = 0; nt < 2; ++nt) {
      v8f c = *cs[mt * 2 + nt];
#pragma unroll
      for (int r = 0; r < 8; ++r)
        sm.osm[mt * 16 + r + hi * 8][nbase + nt * 16 + lm] = f32_to_bf16(c[r]);
    }
  __syncthreads();
  __bf16* obase = attn + (size_t)(b * Cc + h * DHEAD) * Ll + l0;
#pragma unroll
  for (int i = 0; i < 4; ++i) {            // 1024 v8bf units: 32 e x 256 l
    int u = tid + i * 256;
    int e = u >> 5, l8 = (u & 31) * 8;
    *(v8bf*)(obase + (size_t)e * Ll + l8) = *(const v8bf*)&sm.osm[e][l8];
  }
}

// ---------------------------------------------------------------------------
// K7: output projection: out[b,o,l] = sum_c w_out[o,c]*attn[b,c,l] + b_out[o].
// Same 64x256 double-buffered tiling as K2; f32 result into d_out.
// ---------------------------------------------------------------------------
__global__ __launch_bounds__(256) void out_proj(
    const __bf16* __restrict__ attn, const __bf16* __restrict__ wo,
    const float* __restrict__ b_out, float* __restrict__ outp) {
  const int l0 = blockIdx.x * 256;
  const int o0 = blockIdx.y * 64;
  const int b  = blockIdx.z;
  __shared__ __align__(16) __bf16 xT[2][256][40];
  int tid = threadIdx.x, wave = tid >> 5, lane = tid & 31;
  int lm = lane & 15, hi = lane >> 4;
  int mrow = wave & 3, ncol = (wave >> 2) * 128;

  // Staging: 4 v8bf loads per thread per K-step (32c x 256l bf16 tile).
  const __bf16* ap[4]; int sc[4], sl[4];
#pragma unroll
  for (int i = 0; i < 4; ++i) {
    int u = tid + i * 256;                 // 0..1023 v8bf units
    sc[i] = u >> 5;                        // c 0..31
    sl[i] = (u & 31) * 8;                  // l 0,8,..,248
    ap[i] = attn + (size_t)(b * Cc + sc[i]) * Ll + l0 + sl[i];
  }
  const size_t kstride = (size_t)32 * Ll;
  auto stage = [&](int buf) {
#pragma unroll
    for (int i = 0; i < 4; ++i) {
      v8bf v = *(const v8bf*)ap[i];
#pragma unroll
      for (int j = 0; j < 8; ++j) xT[buf][sl[i] + j][sc[i]] = v[j];
      ap[i] += kstride;
    }
  };

  v8f acc[8] = {};
  const __bf16* wrow = wo + (size_t)(o0 + mrow * 16 + lm) * Cc + hi * 8;
  stage(0);
  __syncthreads();
  for (int k0 = 0; k0 < Cc; k0 += 32) {
    const int cur = (k0 >> 5) & 1;
    if (k0 + 32 < Cc) stage(cur ^ 1);
    v16bf afrag = pack16(*(const v8bf*)(wrow + k0), *(const v8bf*)(wrow + k0 + 16));
#pragma unroll
    for (int nt = 0; nt < 8; ++nt) {
      const __bf16* bp = &xT[cur][ncol + nt * 16 + lm][hi * 8];
      v16bf bfrag = pack16(*(const v8bf*)bp, *(const v8bf*)(bp + 16));
      if (nt == 0) acc[nt] = wmma_bf16<false, false>(afrag, bfrag, acc[nt]);
      else         acc[nt] = wmma_bf16<true,  false>(afrag, bfrag, acc[nt]);
    }
    __syncthreads();
  }
#pragma unroll
  for (int nt = 0; nt < 8; ++nt)
#pragma unroll
    for (int r = 0; r < 8; ++r) {
      int o = o0 + mrow * 16 + r + hi * 8;
      int l = l0 + ncol + nt * 16 + lm;
      outp[(size_t)(b * Cc + o) * Ll + l] = acc[nt][r] + b_out[o];
    }
}

// ---------------------------------------------------------------------------
// K8/K9: GroupNorm(1, C): per-batch mean/var over (C,L), then normalize+affine.
// ---------------------------------------------------------------------------
__global__ __launch_bounds__(256) void gn_stats(
    const float* __restrict__ outp, float* __restrict__ stats) {
  int b = blockIdx.y;
  const int chunk4 = (Cc * Ll / 64) / 4;           // 16384 float4 per block
  const float4* p = (const float4*)(outp + (size_t)b * Cc * Ll) +
                    (size_t)blockIdx.x * chunk4;
  float s = 0.f, s2 = 0.f;
  for (int i = threadIdx.x; i < chunk4; i += 256) {
    float4 v = p[i];
    s  += v.x + v.y + v.z + v.w;
    s2 += v.x * v.x + v.y * v.y + v.z * v.z + v.w * v.w;
  }
  __shared__ float r1[256], r2[256];
  int t = threadIdx.x;
  r1[t] = s; r2[t] = s2; __syncthreads();
  for (int st = 128; st; st >>= 1) {
    if (t < st) { r1[t] += r1[t + st]; r2[t] += r2[t + st]; }
    __syncthreads();
  }
  if (t == 0) { atomicAdd(&stats[b * 2], r1[0]); atomicAdd(&stats[b * 2 + 1], r2[0]); }
}

__global__ __launch_bounds__(256) void gn_apply(
    float* __restrict__ outp, const float* __restrict__ stats,
    const float* __restrict__ gw, const float* __restrict__ gb) {
  size_t u = (size_t)blockIdx.x * 256 + threadIdx.x;   // float4 index
  size_t base = u * 4;
  int b = (int)(base / ((size_t)Cc * Ll));
  int o = (int)((base / Ll) % Cc);                      // Ll % 4 == 0 -> same o
  const float N = (float)Cc * (float)Ll;
  float m   = stats[b * 2] / N;
  float var = stats[b * 2 + 1] / N - m * m;
  float rs  = rsqrtf(var + GN_EPS);
  float w = gw[o], bb = gb[o];
  float4 v = *(float4*)(outp + base);
  v.x = (v.x - m) * rs * w + bb;
  v.y = (v.y - m) * rs * w + bb;
  v.z = (v.z - m) * rs * w + bb;
  v.w = (v.w - m) * rs * w + bb;
  *(float4*)(outp + base) = v;
}

// ---------------------------------------------------------------------------
extern "C" void kernel_launch(void* const* d_in, const int* in_sizes, int n_in,
                              void* d_out, int out_size, void* d_ws, size_t ws_size,
                              hipStream_t stream) {
  const float* x     = (const float*)d_in[0];
  const float* w_qkv = (const float*)d_in[1];
  const float* w_out = (const float*)d_in[2];
  const float* b_out = (const float*)d_in[3];
  const float* gn_w  = (const float*)d_in[4];
  const float* gn_b  = (const float*)d_in[5];
  float* outp = (float*)d_out;

  // Workspace carve-up (~270 MB total), 256B-aligned slices.
  char* ws = (char*)d_ws;
  size_t off = 0;
  auto take = [&](size_t bytes) -> char* {
    char* p = ws + off;
    off = (off + bytes + 255) & ~(size_t)255;
    return p;
  };
  __bf16* wq_bf = (__bf16*)take((size_t)O3 * Cc * 2);
  __bf16* wo_bf = (__bf16*)take((size_t)Cc * Cc * 2);
  __bf16* qkv   = (__bf16*)take((size_t)Bx * O3 * Ll * 2);   // 192 MB
  __bf16* attn  = (__bf16*)take((size_t)Bx * Cc * Ll * 2);   //  64 MB
  float*  ctx   = (float*)take((size_t)Bx * HEADS * DHEAD * DHEAD * 4);
  float*  stats = (float*)take((size_t)Bx * 2 * 4);

  hipMemsetAsync(stats, 0, Bx * 2 * sizeof(float), stream);

  cast_weights<<<dim3((O3 * Cc + 255) / 256), 256, 0, stream>>>(w_qkv, w_out, wq_bf, wo_bf);
  qkv_gemm    <<<dim3(Ll / 256, O3 / 64, Bx), 256, 0, stream>>>(x, wq_bf, qkv);
  q_softmax   <<<dim3((Bx * HEADS * Ll) / 256), 256, 0, stream>>>(qkv);
  k_softmax   <<<dim3(Bx * Cc), 256, 0, stream>>>(qkv);
  context_gemm<<<dim3(Bx * HEADS), 256, 0, stream>>>(qkv, ctx);
  attn_apply  <<<dim3(Ll / 256, Bx * HEADS), 256, 0, stream>>>(qkv, ctx, attn);
  out_proj    <<<dim3(Ll / 256, Cc / 64, Bx), 256, 0, stream>>>(attn, wo_bf, b_out, outp);
  gn_stats    <<<dim3(64, Bx), 256, 0, stream>>>(outp, stats);
  gn_apply    <<<dim3((Bx * Cc * Ll) / (256 * 4)), 256, 0, stream>>>(outp, stats, gn_w, gn_b);
}
